// Encoder_28501402976965
// MI455X (gfx1250) — compile-verified, matrix operand order checked by reading, not run on hardware
//
#include <hip/hip_runtime.h>
#include <hip/hip_bf16.h>

// ---------------- problem constants ----------------
constexpr int Bc  = 16;
constexpr int Lc  = 2048;
constexpr int EMBc = 512;
constexpr int NLc = 4;
constexpr int NSc = 16;
constexpr int DIc = 1024;      // EXP * EMB
constexpr int DRc = 32;        // EMB / 16
constexpr int Mrows = Bc * Lc; // 32768

// GEMM tiling: block 256x64, 8 waves as 4(M) x 2(N), wave tile 64x32
constexpr int BMt = 256;
constexpr int BNt = 64;
constexpr int BKt = 32;

// ---------------- vector / fragment types ----------------
typedef __attribute__((ext_vector_type(16))) __bf16 v16bf;
typedef __attribute__((ext_vector_type(8)))  float  v8f;
typedef __attribute__((ext_vector_type(4)))  unsigned int v4u;

union Frag { v16bf bf; v4u u[2]; };
static_assert(sizeof(Frag) == 32, "frag size");

__device__ __forceinline__ unsigned short f2bf(float f) {
  unsigned int u = __float_as_uint(f);
  unsigned int r = u + 0x7FFFu + ((u >> 16) & 1u);   // round-to-nearest-even
  return (unsigned short)(r >> 16);
}
__device__ __forceinline__ float silu(float x) { return x / (1.f + __expf(-x)); }
__device__ __forceinline__ float softplus(float x) {
  return (x > 0.f) ? x + log1pf(__expf(-x)) : log1pf(__expf(x));
}

// LDS byte offset of a __shared__ object (addrspace(3) ptrtoint)
__device__ __forceinline__ unsigned lds_off(const void* p) {
  return (unsigned)(unsigned long long)(const __attribute__((address_space(3))) void*)p;
}
// CDNA5 async memory->LDS copy, 16 bytes per lane, tracked by ASYNCcnt
__device__ __forceinline__ void async_ld_b128(unsigned lofs, const void* gaddr) {
  asm volatile("global_load_async_to_lds_b128 %0, %1, off"
               :: "v"(lofs), "v"(gaddr) : "memory");
}

// ---------------- generic f32 -> bf16 convert ----------------
__global__ void f32_to_bf16_kernel(const float* __restrict__ in,
                                   unsigned short* __restrict__ out, int n) {
  int i = blockIdx.x * blockDim.x + threadIdx.x;
  if (i < n) out[i] = f2bf(in[i]);
}

// ---------------- time2vec + input projection ----------------
__global__ __launch_bounds__(256) void t2v_input_kernel(
    const float* __restrict__ x, const float* __restrict__ lw,
    const float* __restrict__ lb, const float* __restrict__ fw,
    const float* __restrict__ fb, const float* __restrict__ iw,
    const float* __restrict__ ib, float* __restrict__ h) {
  __shared__ float feat[520];
  const int row = blockIdx.x;          // b*L + l
  const int l   = row & (Lc - 1);
  const int tid = threadIdx.x;
  const float* xr = x + (size_t)row * 3;
  const float t = xr[2];
  if (tid == 0) {
    feat[0] = xr[0];
    feat[1] = xr[1];
    float tp = (l == 0) ? 0.f : xr[-1];           // x[row-1][2]
    feat[2] = t - tp;
    feat[3] = t * lw[0] + lb[0];
  }
  for (int i = tid; i < EMBc - 1; i += 256)
    feat[4 + i] = sinf(t * fw[i] + fb[i]);
  __syncthreads();
  float* hr = h + (size_t)row * EMBc;
  for (int e = tid; e < EMBc; e += 256) {
    const float* wrow = iw + (size_t)e * 515;
    float acc = ib[e];
    #pragma unroll 5
    for (int f = 0; f < 515; f++) acc += feat[f] * wrow[f];
    hr[e] = acc;
  }
}

// ---------------- layernorm (f32 in, bf16 out) ----------------
__global__ __launch_bounds__(256) void layernorm_kernel(
    const float* __restrict__ h, const float* __restrict__ w,
    const float* __restrict__ b, unsigned short* __restrict__ outbf) {
  __shared__ float red[256];
  const int row = blockIdx.x;
  const int tid = threadIdx.x;
  const float* hr = h + (size_t)row * EMBc;
  float v0 = hr[tid], v1 = hr[tid + 256];
  red[tid] = v0 + v1;
  __syncthreads();
  for (int s = 128; s > 0; s >>= 1) { if (tid < s) red[tid] += red[tid + s]; __syncthreads(); }
  float mean = red[0] * (1.f / EMBc);
  __syncthreads();
  float d0 = v0 - mean, d1 = v1 - mean;
  red[tid] = d0 * d0 + d1 * d1;
  __syncthreads();
  for (int s = 128; s > 0; s >>= 1) { if (tid < s) red[tid] += red[tid + s]; __syncthreads(); }
  float rstd = rsqrtf(red[0] * (1.f / EMBc) + 1e-5f);
  unsigned short* orow = outbf + (size_t)row * EMBc;
  orow[tid]       = f2bf(d0 * rstd * w[tid]       + b[tid]);
  orow[tid + 256] = f2bf(d1 * rstd * w[tid + 256] + b[tid + 256]);
}

// ---------------- WMMA bf16 GEMM: C[M,N] (+)= A[M,K] * W[N,K]^T ----------------
// Async-to-LDS double buffered, last K-chunk peeled so the hot loop is
// branch-free. Block tile 256x64, wave tile 64x32 (4x2 wmma).
// mode 0: C = val (+bias) ; mode 1: C += val ; mode 2: C = softplus(val + bias)
// Cbf (optional): also store bf16 copy of the result.
__global__ __launch_bounds__(256) void gemm_bf16_kernel(
    const unsigned short* __restrict__ A, int lda,
    const unsigned short* __restrict__ W,
    float* __restrict__ C, int ldc,
    unsigned short* __restrict__ Cbf,
    const float* __restrict__ bias,
    int K, int mode) {
  __shared__ unsigned short As[2][BMt * BKt];   // 2 x 16 KB
  __shared__ unsigned short Ws[2][BNt * BKt];   // 2 x 4 KB
  const int tid  = threadIdx.x;
  const int lane = tid & 31;
  const int wave = tid >> 5;
  const int bm = blockIdx.x * BMt;
  const int bn = blockIdx.y * BNt;
  const int wm = (wave >> 1) * 64;    // 0,64,128,192
  const int wn = (wave & 1) * 32;     // 0,32
  v8f acc[4][2] = {};

  const int mr  = lane & 15;
  const int klo = (lane >> 4) << 3;   // 0 or 8: K-half select per fragment layout

  // per-chunk async staging: A tile 256x32 (4 b128/thread), W tile 64x32 (1)
  auto issue = [&](int par, int k0) {
    #pragma unroll
    for (int i = 0; i < 4; i++) {
      const int c = tid + i * 256;          // 0..1023
      const int r = c >> 2, col = (c & 3) * 8;
      async_ld_b128(lds_off(&As[par][r * BKt + col]),
                    A + (size_t)(bm + r) * lda + k0 + col);
    }
    {
      const int r = tid >> 2, col = (tid & 3) * 8;
      async_ld_b128(lds_off(&Ws[par][r * BKt + col]),
                    W + (size_t)(bn + r) * K + k0 + col);
    }
  };

  auto compute = [&](int par) {
    Frag a[4], bfr[2];
    #pragma unroll
    for (int j = 0; j < 2; j++) {
      const unsigned short* q = &Ws[par][(wn + j * 16 + mr) * BKt];
      bfr[j].u[0] = *(const v4u*)(q + klo);
      bfr[j].u[1] = *(const v4u*)(q + 16 + klo);
    }
    #pragma unroll
    for (int i = 0; i < 4; i++) {
      const unsigned short* p = &As[par][(wm + i * 16 + mr) * BKt];
      a[i].u[0] = *(const v4u*)(p + klo);
      a[i].u[1] = *(const v4u*)(p + 16 + klo);
    }
    #pragma unroll
    for (int i = 0; i < 4; i++)
      #pragma unroll
      for (int j = 0; j < 2; j++)
        acc[i][j] = __builtin_amdgcn_wmma_f32_16x16x32_bf16(
            false, a[i].bf, false, bfr[j].bf, (short)0, acc[i][j], false, false);
  };

  issue(0, 0);
  int par = 0;
  int k0 = 0;
  // steady state: unconditionally prefetch the next chunk
  #pragma unroll 1
  for (; k0 + BKt < K; k0 += BKt) {
    issue(par ^ 1, k0 + BKt);
    asm volatile("s_wait_asynccnt 0x5" ::: "memory");  // current chunk landed
    __syncthreads();
    compute(par);
    __syncthreads();   // all waves done reading before this buffer is re-filled
    par ^= 1;
  }
  // peeled last chunk
  asm volatile("s_wait_asynccnt 0x0" ::: "memory");
  __syncthreads();
  compute(par);

  const int nl = lane & 15;
  const int mo = (lane >> 4) * 8;
  #pragma unroll
  for (int ti = 0; ti < 4; ti++)
  #pragma unroll
  for (int tj = 0; tj < 2; tj++) {
    const int col = bn + wn + tj * 16 + nl;
    const float bval = bias ? bias[col] : 0.f;
    #pragma unroll
    for (int r = 0; r < 8; r++) {
      const int row = bm + wm + ti * 16 + mo + r;
      float v = acc[ti][tj][r];
      float* cp = C + (size_t)row * ldc + col;
      float outv;
      if (mode == 1)      outv = *cp + v;
      else if (mode == 2) outv = softplus(v + bval);
      else                outv = v + bval;
      *cp = outv;
      if (Cbf) Cbf[(size_t)row * ldc + col] = f2bf(outv);
    }
  }
}

// ---------------- causal depthwise conv (DC=4) + SiLU ----------------
__global__ void conv_silu_kernel(const float* __restrict__ xz,
                                 const float* __restrict__ cw,
                                 const float* __restrict__ cb,
                                 float* __restrict__ xc,
                                 unsigned short* __restrict__ xcbf, int total) {
  int idx = blockIdx.x * blockDim.x + threadIdx.x;
  if (idx >= total) return;
  const int d   = idx & (DIc - 1);
  const int row = idx >> 10;            // b*L + l
  const int l   = row & (Lc - 1);
  float acc = cb[d];
  const float* w4 = cw + d * 4;
  #pragma unroll
  for (int k = 0; k < 4; k++) {
    int ls = l - 3 + k;
    if (ls >= 0) acc += w4[k] * xz[(size_t)(row - 3 + k) * (2 * DIc) + d];
  }
  float s = silu(acc);
  xc[idx]   = s;
  xcbf[idx] = f2bf(s);
}

// ---------------- selective scan + D-term + SiLU(z) gate ----------------
__global__ __launch_bounds__(256) void scan_kernel(
    const float* __restrict__ dt, const float* __restrict__ xz,
    const float* __restrict__ xc, const float* __restrict__ dbl,
    const float* __restrict__ Alog, const float* __restrict__ Dv,
    unsigned short* __restrict__ ybf) {
  const int id = blockIdx.x * 256 + threadIdx.x;   // 0 .. B*DI-1
  const int b = id >> 10;
  const int d = id & (DIc - 1);
  float Aneg[NSc];
  #pragma unroll
  for (int n = 0; n < NSc; n++) Aneg[n] = -__expf(Alog[d * NSc + n]);
  float s[NSc];
  #pragma unroll
  for (int n = 0; n < NSc; n++) s[n] = 0.f;
  const float Dd = Dv[d];
  for (int l = 0; l < Lc; l++) {
    const size_t row = (size_t)(b * Lc + l);
    const float dtv = dt[row * DIc + d];
    const float xv  = xc[row * DIc + d];
    const float zv  = xz[row * (2 * DIc) + DIc + d];
    const float* bc = dbl + row * 64;
    const float px = dtv * xv;
    float y = 0.f;
    #pragma unroll
    for (int n = 0; n < NSc; n++) {
      s[n] = __expf(dtv * Aneg[n]) * s[n] + px * bc[DRc + n];
      y += s[n] * bc[DRc + NSc + n];
    }
    y = (y + Dd * xv) * silu(zv);
    ybf[row * DIc + d] = f2bf(y);
  }
}

// ---------------- decoder head ----------------
__global__ void decoder_kernel(const float* __restrict__ h,
                               const float* __restrict__ dw,
                               const float* __restrict__ db,
                               float* __restrict__ out) {
  const int id = blockIdx.x * 256 + threadIdx.x;   // 0 .. 2047
  const int bi = id >> 7, j = id & 127;
  const float* hr = h + ((size_t)(bi * Lc + Lc - 1)) * EMBc;
  const float* wr = dw + (size_t)j * EMBc;
  float acc = db[j];
  #pragma unroll 4
  for (int e = 0; e < EMBc; e++) acc += hr[e] * wr[e];
  if (j < 64) out[bi * 64 + j] = acc;
  else        out[1024 + bi * 64 + (j - 64)] = softplus(acc) + 1e-5f;
}

// ---------------- host orchestration ----------------
extern "C" void kernel_launch(void* const* d_in, const int* in_sizes, int n_in,
                              void* d_out, int out_size, void* d_ws, size_t ws_size,
                              hipStream_t stream) {
  const float* x      = (const float*)d_in[0];
  const float* lin_w  = (const float*)d_in[1];
  const float* lin_b  = (const float*)d_in[2];
  const float* freq_w = (const float*)d_in[3];
  const float* freq_b = (const float*)d_in[4];
  const float* inp_w  = (const float*)d_in[5];
  const float* inp_b  = (const float*)d_in[6];
  const float* ln_w   = (const float*)d_in[7];
  const float* ln_b   = (const float*)d_in[8];
  const float* ipw    = (const float*)d_in[9];
  const float* conv_w = (const float*)d_in[10];
  const float* conv_b = (const float*)d_in[11];
  const float* xpw    = (const float*)d_in[12];
  const float* dtw    = (const float*)d_in[13];
  const float* dtb    = (const float*)d_in[14];
  const float* Alog   = (const float*)d_in[15];
  const float* Dp     = (const float*)d_in[16];
  const float* opw    = (const float*)d_in[17];
  const float* dec_w  = (const float*)d_in[18];
  const float* dec_b  = (const float*)d_in[19];
  float* out = (float*)d_out;

  char* ws = (char*)d_ws;
  size_t off = 0;
  auto carve = [&](size_t bytes) -> char* {
    char* p = ws + off;
    off += (bytes + 255) & ~(size_t)255;
    return p;
  };
  float*          h      = (float*)carve((size_t)Mrows * EMBc * 4);
  unsigned short* hn_bf  = (unsigned short*)carve((size_t)Mrows * EMBc * 2);
  float*          xz     = (float*)carve((size_t)Mrows * 2 * DIc * 4);
  float*          xc     = (float*)carve((size_t)Mrows * DIc * 4);
  unsigned short* xc_bf  = (unsigned short*)carve((size_t)Mrows * DIc * 2);
  float*          dbl    = (float*)carve((size_t)Mrows * 64 * 4);
  unsigned short* dbl_bf = (unsigned short*)carve((size_t)Mrows * 64 * 2);
  float*          dtbuf  = (float*)carve((size_t)Mrows * DIc * 4);
  unsigned short* y_bf   = (unsigned short*)carve((size_t)Mrows * DIc * 2);
  unsigned short* ipw_bf = (unsigned short*)carve((size_t)NLc * 2 * DIc * EMBc * 2);
  unsigned short* xpw_bf = (unsigned short*)carve((size_t)NLc * 64 * DIc * 2);
  unsigned short* dtw_bf = (unsigned short*)carve((size_t)NLc * DIc * DRc * 2);
  unsigned short* opw_bf = (unsigned short*)carve((size_t)NLc * EMBc * DIc * 2);

  auto cvt = [&](const float* src, unsigned short* dst, int n) {
    f32_to_bf16_kernel<<<(n + 255) / 256, 256, 0, stream>>>(src, dst, n);
  };
  cvt(ipw, ipw_bf, NLc * 2 * DIc * EMBc);
  cvt(xpw, xpw_bf, NLc * 64 * DIc);
  cvt(dtw, dtw_bf, NLc * DIc * DRc);
  cvt(opw, opw_bf, NLc * EMBc * DIc);

  t2v_input_kernel<<<Mrows, 256, 0, stream>>>(x, lin_w, lin_b, freq_w, freq_b,
                                              inp_w, inp_b, h);

  auto gemm = [&](const unsigned short* A, int lda, const unsigned short* W,
                  float* Cout, int ldc, unsigned short* Cbf,
                  const float* bias, int N, int K, int mode) {
    dim3 grid(Mrows / BMt, N / BNt);
    gemm_bf16_kernel<<<grid, 256, 0, stream>>>(A, lda, W, Cout, ldc, Cbf, bias, K, mode);
  };

  const int totalXC = Mrows * DIc;
  for (int l = 0; l < NLc; l++) {
    layernorm_kernel<<<Mrows, 256, 0, stream>>>(h, ln_w + l * EMBc, ln_b + l * EMBc, hn_bf);
    // xz = hn @ in_proj^T                       (M x 2048, K=512)
    gemm(hn_bf, EMBc, ipw_bf + (size_t)l * 2 * DIc * EMBc, xz, 2 * DIc, nullptr,
         nullptr, 2 * DIc, EMBc, 0);
    // depthwise causal conv + SiLU
    conv_silu_kernel<<<(totalXC + 255) / 256, 256, 0, stream>>>(
        xz, conv_w + l * DIc * 4, conv_b + l * DIc, xc, xc_bf, totalXC);
    // dbl = xc @ x_proj^T (M x 64, K=1024), fused bf16 copy for the dt GEMM
    gemm(xc_bf, DIc, xpw_bf + (size_t)l * 64 * DIc, dbl, 64, dbl_bf,
         nullptr, 64, DIc, 0);
    // dt = softplus(dbl[:, :32] @ dt_w^T + dt_b)  (M x 1024, K=32)
    gemm(dbl_bf, 64, dtw_bf + (size_t)l * DIc * DRc, dtbuf, DIc, nullptr,
         dtb + l * DIc, DIc, DRc, 2);
    // selective scan + D*xc + SiLU(z) gate -> y (bf16)
    scan_kernel<<<(Bc * DIc) / 256, 256, 0, stream>>>(
        dtbuf, xz, xc, dbl, Alog + (size_t)l * DIc * NSc, Dp + l * DIc, y_bf);
    // h += y @ out_proj^T                        (M x 512, K=1024)
    gemm(y_bf, DIc, opw_bf + (size_t)l * EMBc * DIc, h, EMBc, nullptr,
         nullptr, EMBc, DIc, 1);
  }

  decoder_kernel<<<(Bc * 128) / 256, 256, 0, stream>>>(h, dec_w, dec_b, out);
}
